// IterativeGaussianProcess2_8744553415328
// MI455X (gfx1250) — matrix operand order repaired and checked by reading, not compiled, not run on hardware
//
#include <hip/hip_runtime.h>
#include <math.h>

// ---------------------------------------------------------------------------
// Types for CDNA5 WMMA (wave32): v_wmma_f32_16x16x32_bf16
// ---------------------------------------------------------------------------
typedef __attribute__((ext_vector_type(8)))  __bf16 bf16x8;
typedef __attribute__((ext_vector_type(16))) __bf16 v16bf;
typedef __attribute__((ext_vector_type(8)))  float  v8f;
typedef unsigned short u16;

// Problem sizes (match reference)
#define NTRAIN 8192
#define DFEAT  64
#define MTEST  1024
#define NPROBE 16
#define CG_ITERS 50
#define LANCZOS_RANK 64

// Scalar slot indices in the sc[] f32 scratch area
#define I_LS2INV 0
#define I_OUTS   1
#define I_S2     2
#define I_RHSN   3
#define I_SUMQ   4
#define I_BETAC  5
#define I_RS     16   // 16 slots
#define I_PAP    32   // 16 slots
#define I_RSN    48   // 16 slots
#define I_ALPHA  64   // 64 slots
#define I_BETA   128  // 64 slots
#define I_LD     192  // 64 slots
#define I_LE     256  // 63 slots

// ---------------------------------------------------------------------------
// bf16 <-> f32 helpers (round-to-nearest-even), no scalar __bf16 arithmetic
// ---------------------------------------------------------------------------
__device__ __forceinline__ u16 f2bf(float f) {
  unsigned u = __float_as_uint(f);
  unsigned r = (u + 0x7FFFu + ((u >> 16) & 1u)) >> 16;
  return (u16)r;
}
__device__ __forceinline__ float bf2f(u16 h) {
  return __uint_as_float(((unsigned)h) << 16);
}

// ---------------------------------------------------------------------------
// Core WMMA 16x16 tile MAC over K, bf16 inputs, f32 accum.
// A: 16 rows, row-major, k-contiguous (lda elems).
// B: 16 "columns" stored as 16 k-contiguous rows (ldb elems) i.e. pre-transposed.
// Per-lane load pattern, contiguous k order shared by A and B:
//   A lane L (m=L&15, half=L>>4): elems 0-15 = K kk+half*16..  (contiguous 16)
//   B lane L (n=L&15, half=L>>4): elems 0-15 = K kk+half*16..  (contiguous 16)
// ---------------------------------------------------------------------------
__device__ __forceinline__ v8f wmma_bf16_tile(const u16* __restrict__ A, long lda,
                                              const u16* __restrict__ B, long ldb,
                                              int ksteps, v8f c) {
  const int lane = threadIdx.x & 31;
  const int half = lane >> 4;
  const int m    = lane & 15;
  const u16* ap = A + (long)m * lda + half * 16;
  const u16* bp = B + (long)m * ldb + half * 16;
  for (int k = 0; k < ksteps; ++k) {
    bf16x8 a0 = *reinterpret_cast<const bf16x8*>(ap);
    bf16x8 a1 = *reinterpret_cast<const bf16x8*>(ap + 8);
    bf16x8 b0 = *reinterpret_cast<const bf16x8*>(bp);
    bf16x8 b1 = *reinterpret_cast<const bf16x8*>(bp + 8);
    __builtin_prefetch((const void*)(ap + 512), 0, 0);
    v16bf av = __builtin_shufflevector(a0, a1, 0,1,2,3,4,5,6,7,8,9,10,11,12,13,14,15);
    v16bf bv = __builtin_shufflevector(b0, b1, 0,1,2,3,4,5,6,7,8,9,10,11,12,13,14,15);
    c = __builtin_amdgcn_wmma_f32_16x16x32_bf16(false, av, false, bv, (short)0, c,
                                                false, false);
    ap += 32; bp += 32;
  }
  return c;
}

// ---------------------------------------------------------------------------
// 4-row-tile variant for the dominant matvec kernel: one wave owns 64 rows
// (4 stacked 16x16 tiles), loads the shared 32x16 B fragment ONCE per k-step
// and feeds 4 back-to-back WMMAs. 2.5 loads per WMMA instead of 4, and the
// B panel traffic drops 4x.
// ---------------------------------------------------------------------------
__device__ __forceinline__ void wmma_bf16_tile4(const u16* __restrict__ A, long lda,
                                                const u16* __restrict__ B, long ldb,
                                                int ksteps, v8f c[4]) {
  const int lane = threadIdx.x & 31;
  const int half = lane >> 4;
  const int m    = lane & 15;
  const u16* ap = A + (long)m * lda + half * 16;
  const u16* bp = B + (long)m * ldb + half * 16;
  const long tstride = 16 * lda;   // 16 rows between stacked tiles
  for (int k = 0; k < ksteps; ++k) {
    bf16x8 b0 = *reinterpret_cast<const bf16x8*>(bp);
    bf16x8 b1 = *reinterpret_cast<const bf16x8*>(bp + 8);
    v16bf bv = __builtin_shufflevector(b0, b1, 0,1,2,3,4,5,6,7,8,9,10,11,12,13,14,15);
#pragma unroll
    for (int t = 0; t < 4; ++t) {
      const u16* at = ap + (long)t * tstride;
      bf16x8 a0 = *reinterpret_cast<const bf16x8*>(at);
      bf16x8 a1 = *reinterpret_cast<const bf16x8*>(at + 8);
      __builtin_prefetch((const void*)(at + 512), 0, 0);
      v16bf av = __builtin_shufflevector(a0, a1, 0,1,2,3,4,5,6,7,8,9,10,11,12,13,14,15);
      c[t] = __builtin_amdgcn_wmma_f32_16x16x32_bf16(false, av, false, bv, (short)0,
                                                     c[t], false, false);
    }
    ap += 32; bp += 32;
  }
}

// ---------------------------------------------------------------------------
// 1. Scalars: softplus transforms
// ---------------------------------------------------------------------------
__device__ __forceinline__ float softplus_f(float u) {
  return (u > 20.f) ? u : log1pf(expf(u));
}
__global__ void k_prep(const float* rn, const float* rls, const float* ros, float* sc) {
  float ls   = 1e-3f + softplus_f(rls[0]);
  float outs = 1e-3f + softplus_f(ros[0]);
  float nz   = 1e-3f + softplus_f(rn[0]);
  sc[I_LS2INV] = 1.f / (ls * ls);
  sc[I_OUTS]   = outs;
  sc[I_S2]     = nz * nz;
}

// ---------------------------------------------------------------------------
// 2. Convert X_train/X_test to bf16, compute row sum-of-squares. 1 thread/row.
// ---------------------------------------------------------------------------
__global__ __launch_bounds__(256) void k_convx2(const float* __restrict__ Xtr,
                                                const float* __restrict__ Xte,
                                                u16* __restrict__ Xtrb, u16* __restrict__ Xteb,
                                                float* __restrict__ x2tr, float* __restrict__ x2te) {
  int r = blockIdx.x * blockDim.x + threadIdx.x;
  if (r >= NTRAIN + MTEST) return;
  const float* src; u16* dst; float* x2; int rr;
  if (r < NTRAIN) { src = Xtr + (size_t)r * DFEAT; dst = Xtrb + (size_t)r * DFEAT; x2 = x2tr; rr = r; }
  else            { rr = r - NTRAIN; src = Xte + (size_t)rr * DFEAT; dst = Xteb + (size_t)rr * DFEAT; x2 = x2te; }
  float s = 0.f;
  for (int d = 0; d < DFEAT; ++d) { float f = src[d]; s += f * f; dst[d] = f2bf(f); }
  x2[rr] = s;
}

// ---------------------------------------------------------------------------
// 3. ||y||  (single block)
// ---------------------------------------------------------------------------
__global__ __launch_bounds__(1024) void k_ynorm(const float* __restrict__ y, float* sc) {
  __shared__ float red[1024];
  int t = threadIdx.x; float s = 0.f;
  for (int i = t; i < NTRAIN; i += 1024) { float v = y[i]; s += v * v; }
  red[t] = s; __syncthreads();
  for (int o = 512; o; o >>= 1) { if (t < o) red[t] += red[t + o]; __syncthreads(); }
  if (t == 0) { float n = sqrtf(red[0]); sc[I_RHSN] = (n < 1e-10f) ? 1.f : n; }
}

// ---------------------------------------------------------------------------
// 4. CG init: col-major 16 x 8192 vectors. Col 0 = y/||y||, cols 1..15 = 0.
// ---------------------------------------------------------------------------
__global__ __launch_bounds__(1024) void k_cginit(const float* __restrict__ y, const float* sc,
                                                 float* X, float* R, float* P, u16* PbT,
                                                 float* scw) {
  int t = threadIdx.x;
  float rn = sc[I_RHSN];
  for (int i = t; i < NTRAIN; i += 1024) {
    float v = y[i] / rn;
    for (int c = 0; c < 16; ++c) {
      float val = (c == 0) ? v : 0.f;
      size_t idx = (size_t)c * NTRAIN + i;
      X[idx] = 0.f; R[idx] = val; P[idx] = val;
      PbT[idx] = (c == 0) ? f2bf(v) : (u16)0;
    }
  }
  if (t < 16) { scw[I_RS + t] = (t == 0) ? 1.f : 0.f; }
}

// ---------------------------------------------------------------------------
// 5. RBF build: out(bf16) = outs * exp(-0.5*max(x2a+x2b-2*A·B,0)*ls2inv)
//    One 16x16 tile per wave; D=64 => 2 WMMA k-steps. Optional transposed store.
// ---------------------------------------------------------------------------
__global__ __launch_bounds__(256) void k_rbf(const u16* __restrict__ Xa, const u16* __restrict__ Xb,
                                             const float* __restrict__ x2a, const float* __restrict__ x2b,
                                             u16* __restrict__ out, int nti, int ntj, long ldo,
                                             int transpose, const float* sc) {
  int wid = threadIdx.x >> 5;
  int t = blockIdx.x * 8 + wid;
  if (t >= nti * ntj) return;
  int ti = t / ntj, tj = t % ntj;
  int i0 = ti * 16, j0 = tj * 16;
  v8f c = {};
  c = wmma_bf16_tile(Xa + (size_t)i0 * DFEAT, DFEAT, Xb + (size_t)j0 * DFEAT, DFEAT, 2, c);
  float outs = sc[I_OUTS], ls2 = sc[I_LS2INV];
  int lane = threadIdx.x & 31, n = lane & 15, half = lane >> 4;
  for (int r = 0; r < 8; ++r) {
    int gi = i0 + 8 * half + r, gj = j0 + n;
    float sq = x2a[gi] + x2b[gj] - 2.f * c[r];
    sq = fmaxf(sq, 0.f);
    u16 v = f2bf(outs * expf(-0.5f * sq * ls2));
    if (transpose) out[(size_t)gj * ldo + gi] = v;
    else           out[(size_t)gi * ldo + gj] = v;
  }
}

// ---------------------------------------------------------------------------
// 6. Matvec/GEMM: Out(16x8192 col-major) = K(bf16 8192x8192) @ PbT(16x8192)^T
//    One wave per 64 rows (4 stacked tiles, shared B fragment).
//    mode 0 (CG):      + s2 * P            (diagonal noise on the solve cols)
//    mode 1 (Lanczos): + s2 * sum(q)       (Kd = K + s2 elementwise rank-1 term)
// ---------------------------------------------------------------------------
__global__ __launch_bounds__(256) void k_gemm(const u16* __restrict__ Kb, const u16* __restrict__ PbT,
                                              const float* __restrict__ Pf, float* __restrict__ Out,
                                              const float* sc, int mode) {
  int wid = threadIdx.x >> 5;
  int task = blockIdx.x * 8 + wid;          // 128 tasks x 64 rows = 8192 rows
  int row0 = task * 64;
  v8f c[4] = {v8f{}, v8f{}, v8f{}, v8f{}};
  wmma_bf16_tile4(Kb + (size_t)row0 * NTRAIN, NTRAIN, PbT, NTRAIN, NTRAIN / 32, c);
  float s2 = sc[I_S2];
  float bq = (mode == 1) ? s2 * sc[I_SUMQ] : 0.f;
  int lane = threadIdx.x & 31, n = lane & 15, half = lane >> 4;
#pragma unroll
  for (int t = 0; t < 4; ++t) {
    for (int r = 0; r < 8; ++r) {
      int row = row0 + t * 16 + 8 * half + r;
      size_t idx = (size_t)n * NTRAIN + row;
      float v = c[t][r];
      if (mode == 0) v += s2 * Pf[idx]; else v += bq;
      Out[idx] = v;
    }
  }
}

// ---------------------------------------------------------------------------
// 7/8/9. CG reductions + updates. One block per column (16 blocks x 256).
// ---------------------------------------------------------------------------
__global__ __launch_bounds__(256) void k_pap(const float* __restrict__ P, const float* __restrict__ Ap,
                                             float* sc) {
  __shared__ float red[256];
  int c = blockIdx.x, t = threadIdx.x;
  const float* p = P + (size_t)c * NTRAIN;
  const float* a = Ap + (size_t)c * NTRAIN;
  float s = 0.f;
  for (int i = t; i < NTRAIN; i += 256) s += p[i] * a[i];
  red[t] = s; __syncthreads();
  for (int o = 128; o; o >>= 1) { if (t < o) red[t] += red[t + o]; __syncthreads(); }
  if (t == 0) sc[I_PAP + c] = red[0];
}

__global__ __launch_bounds__(256) void k_xr(float* __restrict__ X, float* __restrict__ R,
                                            const float* __restrict__ P, const float* __restrict__ Ap,
                                            float* sc) {
  __shared__ float red[256];
  int c = blockIdx.x, t = threadIdx.x;
  float a = sc[I_RS + c] / (sc[I_PAP + c] + 1e-20f);
  size_t base = (size_t)c * NTRAIN;
  float s = 0.f;
  for (int i = t; i < NTRAIN; i += 256) {
    float p = P[base + i], ap = Ap[base + i];
    X[base + i] += a * p;
    float r = R[base + i] - a * ap;
    R[base + i] = r;
    s += r * r;
  }
  red[t] = s; __syncthreads();
  for (int o = 128; o; o >>= 1) { if (t < o) red[t] += red[t + o]; __syncthreads(); }
  if (t == 0) sc[I_RSN + c] = red[0];
}

__global__ __launch_bounds__(256) void k_pupd(float* __restrict__ P, const float* __restrict__ R,
                                              u16* __restrict__ PbT, float* sc) {
  int c = blockIdx.x, t = threadIdx.x;
  float bet = sc[I_RSN + c] / (sc[I_RS + c] + 1e-20f);
  size_t base = (size_t)c * NTRAIN;
  for (int i = t; i < NTRAIN; i += 256) {
    float p = R[base + i] + bet * P[base + i];
    P[base + i] = p;
    PbT[base + i] = f2bf(p);
  }
  __syncthreads();
  if (t == 0) sc[I_RS + c] = sc[I_RSN + c];
}

// ---------------------------------------------------------------------------
// 10/11/12. Lanczos on Kd. Single-block kernels for the serial vector work.
// ---------------------------------------------------------------------------
__global__ __launch_bounds__(1024) void k_laninit(const float* __restrict__ probes,
                                                  float* __restrict__ q, float* __restrict__ qprev,
                                                  u16* __restrict__ PbT, float* sc) {
  __shared__ float red[1024];
  int t = threadIdx.x;
  float s = 0.f;
  for (int i = t; i < NTRAIN; i += 1024) { float v = probes[(size_t)i * NPROBE]; s += v * v; }
  red[t] = s; __syncthreads();
  for (int o = 512; o; o >>= 1) { if (t < o) red[t] += red[t + o]; __syncthreads(); }
  float inv = 1.f / (sqrtf(red[0]) + 1e-10f);
  for (int i = t; i < NTRAIN; i += 1024) {
    q[i] = probes[(size_t)i * NPROBE] * inv;
    qprev[i] = 0.f;
  }
  for (size_t i = t + NTRAIN; i < (size_t)16 * NTRAIN; i += 1024) PbT[i] = 0;
  if (t == 0) sc[I_BETAC] = 0.f;
}

__global__ __launch_bounds__(1024) void k_lanpre(const float* __restrict__ q,
                                                 u16* __restrict__ PbT, float* sc) {
  __shared__ float red[1024];
  int t = threadIdx.x;
  float s = 0.f;
  for (int i = t; i < NTRAIN; i += 1024) { float v = q[i]; s += v; PbT[i] = f2bf(v); }
  red[t] = s; __syncthreads();
  for (int o = 512; o; o >>= 1) { if (t < o) red[t] += red[t + o]; __syncthreads(); }
  if (t == 0) sc[I_SUMQ] = red[0];
}

__global__ __launch_bounds__(1024) void k_lanstep(const float* __restrict__ W,
                                                  float* __restrict__ q, float* __restrict__ qprev,
                                                  float* __restrict__ Qm, float* sc, int it) {
  __shared__ float red[1024];
  int t = threadIdx.x;
  float betaC = sc[I_BETAC];
  float wv[8], ql[8];
  float dot = 0.f;
  for (int j = 0; j < 8; ++j) {
    int i = t + 1024 * j;
    float w = W[i] - betaC * qprev[i];
    float qv = q[i];
    wv[j] = w; ql[j] = qv; dot += qv * w;
  }
  red[t] = dot; __syncthreads();
  for (int o = 512; o; o >>= 1) { if (t < o) red[t] += red[t + o]; __syncthreads(); }
  float alpha = red[0]; __syncthreads();
  float nr = 0.f;
  for (int j = 0; j < 8; ++j) { wv[j] -= alpha * ql[j]; nr += wv[j] * wv[j]; }
  red[t] = nr; __syncthreads();
  for (int o = 512; o; o >>= 1) { if (t < o) red[t] += red[t + o]; __syncthreads(); }
  float bn = sqrtf(red[0]);
  float inv = 1.f / (bn + 1e-10f);
  for (int j = 0; j < 8; ++j) {
    int i = t + 1024 * j;
    Qm[(size_t)i * LANCZOS_RANK + it] = ql[j];   // Q row-major (N x rank)
    qprev[i] = ql[j];
    q[i] = wv[j] * inv;
  }
  if (t == 0) { sc[I_ALPHA + it] = alpha; sc[I_BETA + it] = bn; sc[I_BETAC] = bn; }
}

// ---------------------------------------------------------------------------
// 13. Cholesky of tridiagonal T + jitter -> bidiagonal L (ld, le). 1 thread.
// ---------------------------------------------------------------------------
__global__ void k_chol(float* sc) {
  float lk = sqrtf(fmaxf(sc[I_ALPHA] + 1e-6f, 1e-12f));
  sc[I_LD] = lk;
  for (int k = 1; k < LANCZOS_RANK; ++k) {
    float e = sc[I_BETA + k - 1] / lk;
    sc[I_LE + k - 1] = e;
    lk = sqrtf(fmaxf(sc[I_ALPHA + k] + 1e-6f - e * e, 1e-12f));
    sc[I_LD + k] = lk;
  }
}

// ---------------------------------------------------------------------------
// 14. F_t = L^{-1} Q^T via 2-term forward recurrence, stored bf16 (64 x 8192).
// ---------------------------------------------------------------------------
__global__ __launch_bounds__(256) void k_fsolve(const float* __restrict__ Qm, const float* sc,
                                                u16* __restrict__ Fb) {
  int i = blockIdx.x * 256 + threadIdx.x;
  if (i >= NTRAIN) return;
  float zp = 0.f;
  for (int r = 0; r < LANCZOS_RANK; ++r) {
    float x = Qm[(size_t)i * LANCZOS_RANK + r];
    if (r) x -= sc[I_LE + r - 1] * zp;
    float z = x / sc[I_LD + r];
    Fb[(size_t)r * NTRAIN + i] = f2bf(z);
    zp = z;
  }
}

// ---------------------------------------------------------------------------
// 15. predictive_mean[m] = sum_i KtT[m][i] * alpha[i];  alpha = X[:,0]*rhs_norm
// ---------------------------------------------------------------------------
__global__ __launch_bounds__(256) void k_mean(const u16* __restrict__ KtT, const float* __restrict__ X0,
                                              const float* sc, float* __restrict__ dout) {
  __shared__ float red[256];
  int wid = threadIdx.x >> 5, lane = threadIdx.x & 31, t = threadIdx.x;
  int m = blockIdx.x * 8 + wid;
  const u16* row = KtT + (size_t)m * NTRAIN;
  float s = 0.f;
  for (int i = lane; i < NTRAIN; i += 32) s += bf2f(row[i]) * X0[i];
  red[t] = s; __syncthreads();
  for (int o = 16; o; o >>= 1) { if ((t & 31) < o) red[t] += red[t + o]; __syncthreads(); }
  if (lane == 0) dout[m] = red[t] * sc[I_RHSN];
}

// ---------------------------------------------------------------------------
// 16. beta = F_t(64x8192) @ K_trans(8192x1024); stored transposed bf16 (1024x64)
// ---------------------------------------------------------------------------
__global__ __launch_bounds__(256) void k_bgemm(const u16* __restrict__ Fb, const u16* __restrict__ KtT,
                                               u16* __restrict__ betaT) {
  int wid = threadIdx.x >> 5;
  int t = blockIdx.x * 8 + wid;        // 256 tiles: 4 (M) x 64 (N)
  int tm = t & 3, tn = t >> 2;
  v8f c = {};
  c = wmma_bf16_tile(Fb + (size_t)tm * 16 * NTRAIN, NTRAIN,
                     KtT + (size_t)tn * 16 * NTRAIN, NTRAIN, NTRAIN / 32, c);
  int lane = threadIdx.x & 31, n = lane & 15, half = lane >> 4;
  for (int r = 0; r < 8; ++r) {
    int gm = tm * 16 + 8 * half + r, gn = tn * 16 + n;
    betaT[(size_t)gn * LANCZOS_RANK + gm] = f2bf(c[r]);
  }
}

// ---------------------------------------------------------------------------
// 17. covariance = K_test - beta^T beta, fused: both Gram-of-Xte and beta^T beta
//     via WMMA (K=64 -> 2 k-steps each). Writes d_out rows 1..1024.
// ---------------------------------------------------------------------------
__global__ __launch_bounds__(256) void k_cov(const u16* __restrict__ Xteb, const u16* __restrict__ betaT,
                                             const float* __restrict__ x2te, const float* sc,
                                             float* __restrict__ dout) {
  int wid = threadIdx.x >> 5;
  int t = blockIdx.x * 8 + wid;        // 4096 tiles: 64 x 64
  int ti = t >> 6, tj = t & 63;
  int i0 = ti * 16, j0 = tj * 16;
  v8f cg = {}, cb = {};
  cg = wmma_bf16_tile(Xteb  + (size_t)i0 * DFEAT,        DFEAT,
                      Xteb  + (size_t)j0 * DFEAT,        DFEAT,        2, cg);
  cb = wmma_bf16_tile(betaT + (size_t)i0 * LANCZOS_RANK, LANCZOS_RANK,
                      betaT + (size_t)j0 * LANCZOS_RANK, LANCZOS_RANK, 2, cb);
  float outs = sc[I_OUTS], ls2 = sc[I_LS2INV];
  int lane = threadIdx.x & 31, n = lane & 15, half = lane >> 4;
  for (int r = 0; r < 8; ++r) {
    int gi = i0 + 8 * half + r, gj = j0 + n;
    float sq = x2te[gi] + x2te[gj] - 2.f * cg[r];
    sq = fmaxf(sq, 0.f);
    float kt = outs * expf(-0.5f * sq * ls2);
    dout[(size_t)(1 + gi) * MTEST + gj] = kt - cb[r];
  }
}

// ---------------------------------------------------------------------------
// Host orchestration
// ---------------------------------------------------------------------------
extern "C" void kernel_launch(void* const* d_in, const int* in_sizes, int n_in,
                              void* d_out, int out_size, void* d_ws, size_t ws_size,
                              hipStream_t stream) {
  const float* Xtr    = (const float*)d_in[0];
  const float* y      = (const float*)d_in[1];
  const float* Xte    = (const float*)d_in[2];
  const float* probes = (const float*)d_in[3];
  const float* rn     = (const float*)d_in[4];
  const float* rls    = (const float*)d_in[5];
  const float* ros    = (const float*)d_in[6];
  float* out = (float*)d_out;

  char* w = (char*)d_ws;
  size_t off = 0;
  auto alloc = [&](size_t bytes) { size_t o = off; off = (off + bytes + 255) & ~(size_t)255; return o; };
  float* sc    = (float*)(w + alloc(4096 * 4));
  u16*   Xtrb  = (u16*)  (w + alloc((size_t)NTRAIN * DFEAT * 2));
  u16*   Xteb  = (u16*)  (w + alloc((size_t)MTEST * DFEAT * 2));
  float* x2tr  = (float*)(w + alloc((size_t)NTRAIN * 4));
  float* x2te  = (float*)(w + alloc((size_t)MTEST * 4));
  u16*   Kb    = (u16*)  (w + alloc((size_t)NTRAIN * NTRAIN * 2));   // 128 MB, L2-resident
  u16*   KtT   = (u16*)  (w + alloc((size_t)MTEST * NTRAIN * 2));    // K_trans^T
  u16*   PbT   = (u16*)  (w + alloc((size_t)16 * NTRAIN * 2));
  float* CGX   = (float*)(w + alloc((size_t)16 * NTRAIN * 4));
  float* CGR   = (float*)(w + alloc((size_t)16 * NTRAIN * 4));
  float* CGP   = (float*)(w + alloc((size_t)16 * NTRAIN * 4));
  float* CGAP  = (float*)(w + alloc((size_t)16 * NTRAIN * 4));
  float* Qm    = (float*)(w + alloc((size_t)NTRAIN * LANCZOS_RANK * 4));
  float* qv    = (float*)(w + alloc((size_t)NTRAIN * 4));
  float* qp    = (float*)(w + alloc((size_t)NTRAIN * 4));
  u16*   Fb    = (u16*)  (w + alloc((size_t)LANCZOS_RANK * NTRAIN * 2));
  u16*   betaT = (u16*)  (w + alloc((size_t)MTEST * LANCZOS_RANK * 2));
  (void)ws_size; (void)in_sizes; (void)n_in; (void)out_size;

  // --- scalars, conversions, norms ---
  k_prep<<<1, 1, 0, stream>>>(rn, rls, ros, sc);
  k_convx2<<<(NTRAIN + MTEST + 255) / 256, 256, 0, stream>>>(Xtr, Xte, Xtrb, Xteb, x2tr, x2te);
  k_ynorm<<<1, 1024, 0, stream>>>(y, sc);
  k_cginit<<<1, 1024, 0, stream>>>(y, sc, CGX, CGR, CGP, PbT, sc);

  // --- build K (bf16, fits in 192MB L2) and K_trans^T ---
  k_rbf<<<(512 * 512) / 8, 256, 0, stream>>>(Xtrb, Xtrb, x2tr, x2tr, Kb, 512, 512, NTRAIN, 0, sc);
  k_rbf<<<(512 * 64) / 8, 256, 0, stream>>>(Xtrb, Xteb, x2tr, x2te, KtT, 512, 64, NTRAIN, 1, sc);

  // --- CG (only the y column matters downstream; columns are independent) ---
  for (int it = 0; it < CG_ITERS; ++it) {
    k_gemm<<<16, 256, 0, stream>>>(Kb, PbT, CGP, CGAP, sc, 0);
    k_pap<<<16, 256, 0, stream>>>(CGP, CGAP, sc);
    k_xr<<<16, 256, 0, stream>>>(CGX, CGR, CGP, CGAP, sc);
    k_pupd<<<16, 256, 0, stream>>>(CGP, CGR, PbT, sc);
  }

  // --- Lanczos on Kd = K + s2 (elementwise) : Kd@q = K@q + s2*sum(q) ---
  k_laninit<<<1, 1024, 0, stream>>>(probes, qv, qp, PbT, sc);
  for (int it = 0; it < LANCZOS_RANK; ++it) {
    k_lanpre<<<1, 1024, 0, stream>>>(qv, PbT, sc);
    k_gemm<<<16, 256, 0, stream>>>(Kb, PbT, CGP, CGAP, sc, 1);
    k_lanstep<<<1, 1024, 0, stream>>>(CGAP, qv, qp, Qm, sc, it);
  }

  // --- bidiagonal Cholesky of tridiagonal T, F_t = L^{-1} Q^T ---
  k_chol<<<1, 1, 0, stream>>>(sc);
  k_fsolve<<<NTRAIN / 256, 256, 0, stream>>>(Qm, sc, Fb);

  // --- outputs ---
  k_mean<<<MTEST / 8, 256, 0, stream>>>(KtT, CGX, sc, out);
  k_bgemm<<<(4 * 64) / 8, 256, 0, stream>>>(Fb, KtT, betaT);
  k_cov<<<(64 * 64) / 8, 256, 0, stream>>>(Xteb, betaT, x2te, sc, out);
}
